// MyLSTM_66614942761197
// MI455X (gfx1250) — compile-verified, matrix operand order checked by reading, not run on hardware
//
#include <hip/hip_runtime.h>
#include <hip/hip_bf16.h>

// ---------------------------------------------------------------------------
// LSTM: T=128, B=128, I=H=1024 on MI455X (gfx1250, wave32, WMMA bf16).
//
//  - Weights packed to one transposed bf16 buffer WT[n'][k], k in [0,2048)
//    (k<1024 -> W_x, else W_h), with INTERLEAVED N:  n' = col*4 + gate
//    (gate order f,i,c,o). A wave's 16x32 WMMA tile covers 8 columns x all
//    4 gates -> LSTM cell update runs in the GEMM epilogue (one kernel
//    launch per timestep, no gates round-trip).
//  - x-part (recurrence-independent) hoisted into one big GEMM Gx = X @ Wx
//    at M=16384 when workspace allows (ws_size branch); the sequential
//    per-step GEMM then only does K=1024.
//  - Inner loops software-pipelined: next K-chunk's fragments are loaded
//    before WMMAs consume the current ones -> partial s_wait_loadcnt
//    instead of full drains.
//  - h ping-pongs between two buffers across steps.
// ---------------------------------------------------------------------------

typedef __attribute__((ext_vector_type(16))) __bf16 v16bf;
typedef __attribute__((ext_vector_type(8)))  float  v8f;

#define T_STEPS 128
#define BATCH   128
#define HID     1024
#define NGATES  4096   // 4*HID, interleaved n' = col*4 + gate
#define KTOT    2048   // x-part (1024) + h-part (1024)

// workspace layout (byte offsets, 256B-aligned)
#define OFF_WT    ((size_t)0)            // 4096*2048*2  = 16 MB
#define OFF_XBF   ((size_t)16777216)     // 16384*1024*2 = 32 MB
#define OFF_HBFA  ((size_t)50331648)     // 128*1024*2   = 256 KB
#define OFF_HBFB  ((size_t)50593792)     // 128*1024*2   = 256 KB
#define OFF_C     ((size_t)50855936)     // 128*1024*4   = 512 KB
#define OFF_BV    ((size_t)51380224)     // 4096*4       = 16 KB
#define OFF_GX    ((size_t)51396608)     // 16384*4096*4 = 256 MB (optional)
#define GX_BYTES  ((size_t)16384 * 4096 * 4)

// ---- helpers ---------------------------------------------------------------

__device__ inline __bf16 f2bf(float f) {
    union { float f; unsigned u; } a; a.f = f;
    unsigned u = a.u;
    unsigned r = (u + 0x7fffu + ((u >> 16) & 1u)) >> 16;   // round-nearest-even
    union { unsigned short s; __bf16 b; } o; o.s = (unsigned short)r;
    return o.b;
}

// CDNA5 16-bit A/B fragment: elements [half*8..+8) and [16+half*8..+8)
// from a K-contiguous row. Two aligned 16B loads -> global_load_b128.
__device__ inline v16bf load_frag16(const __bf16* __restrict__ p, int half) {
    union { uint4 u[2]; v16bf v; } x;
    x.u[0] = *reinterpret_cast<const uint4*>(p + half * 8);
    x.u[1] = *reinterpret_cast<const uint4*>(p + 16 + half * 8);
    return x.v;
}

__device__ inline float sigf(float x) { return 1.0f / (1.0f + __expf(-x)); }

// ---- setup kernels ---------------------------------------------------------

__global__ __launch_bounds__(256) void conv_x_kernel(const float* __restrict__ x,
                                                     __bf16* __restrict__ xb, int n) {
    int i = blockIdx.x * blockDim.x + threadIdx.x;
    if (i < n) xb[i] = f2bf(x[i]);
}

// WT[n'][k]: n' = col*4 + gate (f,i,c,o); k<1024 -> W_x[k][col], else W_h.
__global__ __launch_bounds__(256) void conv_w_kernel(
    const float* __restrict__ Wxf, const float* __restrict__ Wxi,
    const float* __restrict__ Wxc, const float* __restrict__ Wxo,
    const float* __restrict__ Whf, const float* __restrict__ Whi,
    const float* __restrict__ Whc, const float* __restrict__ Who,
    __bf16* __restrict__ WT) {
    int idx = blockIdx.x * blockDim.x + threadIdx.x;   // 0 .. 4096*2048-1
    int k   = idx & (KTOT - 1);
    int n   = idx >> 11;
    int col = n >> 2;
    int g   = n & 3;
    const float* Wx = (g == 0) ? Wxf : (g == 1) ? Wxi : (g == 2) ? Wxc : Wxo;
    const float* Wh = (g == 0) ? Whf : (g == 1) ? Whi : (g == 2) ? Whc : Who;
    float v = (k < HID) ? Wx[(size_t)k * HID + col]
                        : Wh[(size_t)(k - HID) * HID + col];
    WT[idx] = f2bf(v);
}

__global__ __launch_bounds__(256) void pack_bias_kernel(
    const float* __restrict__ bf_, const float* __restrict__ bi_,
    const float* __restrict__ bc_, const float* __restrict__ bo_,
    float* __restrict__ Bv) {
    int i = blockIdx.x * blockDim.x + threadIdx.x;
    if (i < NGATES) {
        int col = i >> 2, g = i & 3;
        const float* b = (g == 0) ? bf_ : (g == 1) ? bi_ : (g == 2) ? bc_ : bo_;
        Bv[i] = b[col];
    }
}

__global__ __launch_bounds__(256) void init_state_kernel(__bf16* __restrict__ hbf,
                                                         float* __restrict__ c) {
    int i = blockIdx.x * blockDim.x + threadIdx.x;
    if (i < BATCH * HID) {
        hbf[i] = f2bf(0.0f);
        c[i]   = 0.0f;
    }
}

// ---- big x-part GEMM (recurrence-independent, full parallelism) ------------
// Gx[(t*B+b)][n'] = sum_{k<1024} X[t*B+b][k] * WT[n'][k]
// Grid (8, 1024), 256 threads; each wave: 16x64 tile, 4 accumulators.
// Software-pipelined: next chunk's 10 b128 loads issued before current WMMAs.

__global__ __launch_bounds__(256) void lstm_xpart_gemm(
    const __bf16* __restrict__ Xbf, const __bf16* __restrict__ WT,
    float* __restrict__ Gx) {
    const int lane = threadIdx.x & 31;
    const int wave = threadIdx.x >> 5;
    const int half = lane >> 4;
    const int l16  = lane & 15;
    const int mtile = blockIdx.y;                   // 0..1023
    const int n0    = blockIdx.x * 512 + wave * 64; // 0..4032

    const __bf16* Arow = Xbf + (size_t)(mtile * 16 + l16) * HID;
    const __bf16* B0   = WT + (size_t)(n0 + l16) * KTOT;
    const __bf16* B1   = B0 + (size_t)16 * KTOT;
    const __bf16* B2   = B0 + (size_t)32 * KTOT;
    const __bf16* B3   = B0 + (size_t)48 * KTOT;

    v8f acc0 = {}, acc1 = {}, acc2 = {}, acc3 = {};

    v16bf a_c  = load_frag16(Arow, half);
    v16bf b0_c = load_frag16(B0, half);
    v16bf b1_c = load_frag16(B1, half);
    v16bf b2_c = load_frag16(B2, half);
    v16bf b3_c = load_frag16(B3, half);

    for (int kc = 0; kc < HID - 32; kc += 32) {
        const int kn = kc + 32;
        v16bf a_n  = load_frag16(Arow + kn, half);
        v16bf b0_n = load_frag16(B0 + kn, half);
        v16bf b1_n = load_frag16(B1 + kn, half);
        v16bf b2_n = load_frag16(B2 + kn, half);
        v16bf b3_n = load_frag16(B3 + kn, half);

        acc0 = __builtin_amdgcn_wmma_f32_16x16x32_bf16(false, a_c, false, b0_c, (short)0, acc0, false, false);
        acc1 = __builtin_amdgcn_wmma_f32_16x16x32_bf16(false, a_c, false, b1_c, (short)0, acc1, false, false);
        acc2 = __builtin_amdgcn_wmma_f32_16x16x32_bf16(false, a_c, false, b2_c, (short)0, acc2, false, false);
        acc3 = __builtin_amdgcn_wmma_f32_16x16x32_bf16(false, a_c, false, b3_c, (short)0, acc3, false, false);

        a_c = a_n; b0_c = b0_n; b1_c = b1_n; b2_c = b2_n; b3_c = b3_n;
    }
    acc0 = __builtin_amdgcn_wmma_f32_16x16x32_bf16(false, a_c, false, b0_c, (short)0, acc0, false, false);
    acc1 = __builtin_amdgcn_wmma_f32_16x16x32_bf16(false, a_c, false, b1_c, (short)0, acc1, false, false);
    acc2 = __builtin_amdgcn_wmma_f32_16x16x32_bf16(false, a_c, false, b2_c, (short)0, acc2, false, false);
    acc3 = __builtin_amdgcn_wmma_f32_16x16x32_bf16(false, a_c, false, b3_c, (short)0, acc3, false, false);

    const int gm = mtile * 16 + half * 8;
#pragma unroll
    for (int r = 0; r < 8; ++r) {
        Gx[(size_t)(gm + r) * NGATES + (n0 + l16)]      = acc0[r];
        Gx[(size_t)(gm + r) * NGATES + (n0 + 16 + l16)] = acc1[r];
        Gx[(size_t)(gm + r) * NGATES + (n0 + 32 + l16)] = acc2[r];
        Gx[(size_t)(gm + r) * NGATES + (n0 + 48 + l16)] = acc3[r];
    }
}

// ---- fused per-step kernel: h-part GEMM + LSTM cell update -----------------
// Grid (32, 8), 128 threads (4 waves -> 256 WGs for WGP spread).
// Wave tile: 16 rows x 32 n' (= 8 cols x 4 gates). kbeg = 1024 (split) or
// 0 (fused, Gx == nullptr). Software-pipelined K loop.

__global__ __launch_bounds__(128) void lstm_step_kernel(
    const __bf16* __restrict__ Xbf,
    const __bf16* __restrict__ Hin,
    __bf16* __restrict__ Hout,
    const __bf16* __restrict__ WT,
    const float* __restrict__ Gx,
    const float* __restrict__ Bv,
    float* __restrict__ c,
    float* __restrict__ out,
    int t, int kbeg) {
    __shared__ float sh[4][16][36];   // pad 36: 16B-aligned rows, bank-spread
    const int lane = threadIdx.x & 31;
    const int wave = threadIdx.x >> 5;
    const int half = lane >> 4;
    const int l16  = lane & 15;
    const int mtile = blockIdx.y;                    // 0..7
    const int n0    = blockIdx.x * 128 + wave * 32;  // 0..4064

    const int arowi = mtile * 16 + l16;
    const __bf16* Axrow = Xbf + ((size_t)t * BATCH + arowi) * HID;
    const __bf16* Ahrow = Hin + (size_t)arowi * HID;
    const __bf16* B0 = WT + (size_t)(n0 + l16) * KTOT;
    const __bf16* B1 = WT + (size_t)(n0 + 16 + l16) * KTOT;

    v8f acc0 = {};
    v8f acc1 = {};

    const __bf16* arow0 = (kbeg < HID) ? (Axrow + kbeg) : (Ahrow + (kbeg - HID));
    v16bf a_c  = load_frag16(arow0, half);
    v16bf b0_c = load_frag16(B0 + kbeg, half);
    v16bf b1_c = load_frag16(B1 + kbeg, half);

    for (int kc = kbeg; kc < KTOT - 32; kc += 32) {
        const int kn = kc + 32;
        const __bf16* arow = (kn < HID) ? (Axrow + kn) : (Ahrow + (kn - HID));
        v16bf a_n  = load_frag16(arow, half);
        v16bf b0_n = load_frag16(B0 + kn, half);
        v16bf b1_n = load_frag16(B1 + kn, half);

        acc0 = __builtin_amdgcn_wmma_f32_16x16x32_bf16(false, a_c, false, b0_c, (short)0, acc0, false, false);
        acc1 = __builtin_amdgcn_wmma_f32_16x16x32_bf16(false, a_c, false, b1_c, (short)0, acc1, false, false);

        a_c = a_n; b0_c = b0_n; b1_c = b1_n;
    }
    acc0 = __builtin_amdgcn_wmma_f32_16x16x32_bf16(false, a_c, false, b0_c, (short)0, acc0, false, false);
    acc1 = __builtin_amdgcn_wmma_f32_16x16x32_bf16(false, a_c, false, b1_c, (short)0, acc1, false, false);

    // Stage C tiles to LDS: row m = r + 8*half, col-in-tile = j*16 + l16.
#pragma unroll
    for (int r = 0; r < 8; ++r) {
        sh[wave][half * 8 + r][l16]      = acc0[r];
        sh[wave][half * 8 + r][16 + l16] = acc1[r];
    }
    __syncthreads();

    // Each lane handles 4 (row,col) cells; gate quad is contiguous (b128).
    const int c0 = n0 >> 2;                         // first of 8 cols for this wave
    const size_t obase = (size_t)t * BATCH * HID;
#pragma unroll
    for (int qq = 0; qq < 4; ++qq) {
        int q  = lane + 32 * qq;                    // 0..127
        int m  = q >> 3;                            // 0..15
        int cl = q & 7;                             // 0..7
        int row = mtile * 16 + m;
        int col = c0 + cl;

        float4 g  = *reinterpret_cast<const float4*>(&sh[wave][m][cl * 4]);
        float4 bq = *reinterpret_cast<const float4*>(&Bv[(size_t)col * 4]);
        float pf = g.x + bq.x, pi = g.y + bq.y, pc = g.z + bq.z, po = g.w + bq.w;
        if (Gx) {
            float4 gx = *reinterpret_cast<const float4*>(
                &Gx[((size_t)t * BATCH + row) * NGATES + (size_t)col * 4]);
            pf += gx.x; pi += gx.y; pc += gx.z; po += gx.w;
        }
        float fg = sigf(pf);
        float ig = sigf(pi);
        float cg = tanhf(pc);
        float og = sigf(po);

        size_t ci = (size_t)row * HID + col;
        float cn = fg * c[ci] + ig * cg;
        float hn = og * tanhf(cn);

        c[ci]    = cn;
        Hout[ci] = f2bf(hn);
        out[obase + ci] = hn;
        if (t == T_STEPS - 1) {
            size_t base = (size_t)T_STEPS * BATCH * HID;
            out[base + ci] = hn;                         // h_T
            out[base + (size_t)BATCH * HID + ci] = cn;   // c_T
        }
    }
}

// ---- launcher ---------------------------------------------------------------

extern "C" void kernel_launch(void* const* d_in, const int* in_sizes, int n_in,
                              void* d_out, int out_size, void* d_ws, size_t ws_size,
                              hipStream_t stream) {
    const float* x   = (const float*)d_in[0];
    const float* Wxf = (const float*)d_in[1];
    const float* Whf = (const float*)d_in[2];
    const float* bfp = (const float*)d_in[3];
    const float* Wxi = (const float*)d_in[4];
    const float* Whi = (const float*)d_in[5];
    const float* bip = (const float*)d_in[6];
    const float* Wxc = (const float*)d_in[7];
    const float* Whc = (const float*)d_in[8];
    const float* bcp = (const float*)d_in[9];
    const float* Wxo = (const float*)d_in[10];
    const float* Who = (const float*)d_in[11];
    const float* bop = (const float*)d_in[12];

    float* out = (float*)d_out;
    char*  ws  = (char*)d_ws;
    __bf16* WT   = (__bf16*)(ws + OFF_WT);
    __bf16* Xbf  = (__bf16*)(ws + OFF_XBF);
    __bf16* HbfA = (__bf16*)(ws + OFF_HBFA);
    __bf16* HbfB = (__bf16*)(ws + OFF_HBFB);
    float*  c    = (float*)(ws + OFF_C);
    float*  Bv   = (float*)(ws + OFF_BV);
    float*  Gx   = (float*)(ws + OFF_GX);

    const bool split = (ws_size >= OFF_GX + GX_BYTES);

    const int nx = T_STEPS * BATCH * HID;            // 16,777,216
    conv_x_kernel<<<(nx + 255) / 256, 256, 0, stream>>>(x, Xbf, nx);
    conv_w_kernel<<<(NGATES * KTOT) / 256, 256, 0, stream>>>(
        Wxf, Wxi, Wxc, Wxo, Whf, Whi, Whc, Who, WT);
    pack_bias_kernel<<<NGATES / 256, 256, 0, stream>>>(bfp, bip, bcp, bop, Bv);
    init_state_kernel<<<(BATCH * HID) / 256, 256, 0, stream>>>(HbfA, c);

    if (split) {
        lstm_xpart_gemm<<<dim3(8, 1024), 256, 0, stream>>>(Xbf, WT, Gx);
    }

    for (int t = 0; t < T_STEPS; ++t) {
        const __bf16* hin  = (t & 1) ? HbfB : HbfA;
        __bf16*       hout = (t & 1) ? HbfA : HbfB;
        lstm_step_kernel<<<dim3(32, 8), 128, 0, stream>>>(
            Xbf, hin, hout, WT, split ? Gx : nullptr, Bv, c, out,
            t, split ? HID : 0);
    }
}
